// NonLocalblock_41798621724719
// MI455X (gfx1250) — matrix operand order repaired and checked by reading, not compiled, not run on hardware
//
#include <hip/hip_runtime.h>
#include <hip/hip_bf16.h>

// ---------------------------------------------------------------------------
// Types for CDNA5 WMMA (wave32): 16x16x32 bf16 -> f32
// ---------------------------------------------------------------------------
typedef __attribute__((ext_vector_type(16))) __bf16 v16bf;
typedef __attribute__((ext_vector_type(8)))  __bf16 v8bf;
typedef __attribute__((ext_vector_type(8)))  float  v8f;
typedef __attribute__((ext_vector_type(4)))  float  v4f;

__device__ __forceinline__ __bf16 f2bf(float x) {
    union { float f; unsigned u; } v; v.f = x;
    unsigned r = v.u + 0x7FFFu + ((v.u >> 16) & 1u);   // round-to-nearest-even
    unsigned short h = (unsigned short)(r >> 16);
    __bf16 out; __builtin_memcpy(&out, &h, 2);
    return out;
}

__device__ __forceinline__ v8f wmma_bf16(const v16bf& a, const v16bf& b, v8f c) {
    return __builtin_amdgcn_wmma_f32_16x16x32_bf16(
               false, a, false, b, (short)0, c, false, false);
}

#define TBLK 256
#define BM 64
#define BN 64
#define BK 32
#define LDK (BK + 16)   // 48 elems = 96B row stride: 16B-aligned vector rows

// ---------------------------------------------------------------------------
// Generic batched WMMA GEMM:  C[bz] = scale * A[bz] x B[bz]
//   LAYA = 0: A[m*K + k] (MK)  -> staged with global_load_async_to_lds_b128
//   LAYA = 1: A[k*M + m] (KM)  -> register-staged + LDS transpose scatter
//   LAYB = 0: B[k*N + n] (KN)  -> register-staged + LDS transpose scatter
//   LAYB = 1: B[n*K + k] (NK)  -> register-staged direct copy
//   BF32 : B operand is f32 in memory (converted to bf16 while staging)
//   OUTBF: store bf16, else f32
// 256 threads = 8 waves; block tile 64x64x32; wave tile 16(M) x 32(N):
// two WMMAs per K-step sharing the A fragment. Double-buffered LDS,
// async/registered staging, one barrier per K-step.
// ---------------------------------------------------------------------------
template<int LAYA, int LAYB, bool BF32, bool OUTBF>
__global__ __launch_bounds__(TBLK)
void gemm_wmma_bf16(const __bf16* __restrict__ Ag, const void* __restrict__ Bg,
                    float* __restrict__ Cf, __bf16* __restrict__ Cb,
                    int M, int N, int K, long sA, long sB, long sC, float scale)
{
    // lsA is declared first -> sits at LDS offset 0 of the group segment.
    __shared__ __bf16 lsA[2][BM][LDK];
    __shared__ __bf16 lsB[2][BN][LDK];

    const int tid = threadIdx.x;
    const int bz  = blockIdx.z;
    const int m0  = blockIdx.y * BM;
    const int n0  = blockIdx.x * BN;

    const __bf16* A   = Ag + (long)bz * sA;
    const __bf16* B16 = (const __bf16*)Bg + (BF32 ? 0 : (long)bz * sB);
    const float*  B32 = (const float*)Bg + (BF32 ? (long)bz * sB : 0);

    // ---- staging thread mapping: 2048 bf16 per operand, 8 per thread ----
    int am, ak;                       // A tile coords
    if (LAYA == 0) { am = (tid * 8) / BK; ak = (tid * 8) % BK; }   // k-contig
    else           { ak = (tid * 8) / BM; am = (tid * 8) % BM; }   // m-contig
    int bn_, bk_;                     // B tile coords
    if (LAYB == 0) { bk_ = (tid * 8) / BN; bn_ = (tid * 8) % BN; } // n-contig
    else           { bn_ = (tid * 8) / BK; bk_ = (tid * 8) % BK; } // k-contig

    // LDS byte address of this thread's A staging slot (group segment base = 0)
    const unsigned ldsA_elem = (unsigned)(am * LDK + ak);
    const unsigned ldsA_bufstride = (unsigned)(BM * LDK * 2);

    // async A staging: 16B contiguous global -> 16B contiguous LDS, ASYNCcnt
    auto asyncStageA = [&](int buf, int k0) {
        unsigned lds = (unsigned)buf * ldsA_bufstride + ldsA_elem * 2u;
        unsigned long long g =
            (unsigned long long)(A + (long)(m0 + am) * K + (k0 + ak));
        asm volatile("global_load_async_to_lds_b128 %0, %1, off"
                     :: "v"(lds), "v"(g) : "memory");
    };
    auto globA = [&](int k0) -> v8bf {     // LAYA==1 path
        return *(const v8bf*)(A + (long)(k0 + ak) * M + (m0 + am));
    };
    auto stageA = [&](int buf, v8bf r) {   // LAYA==1 path: transpose scatter
        #pragma unroll
        for (int j = 0; j < 8; ++j) lsA[buf][am + j][ak] = r[j];
    };
    auto globB = [&](int k0) -> v8bf {
        long off = (LAYB == 0) ? (long)(k0 + bk_) * N + (n0 + bn_)
                               : (long)(n0 + bn_) * K + (k0 + bk_);
        if (BF32) {
            v4f f0 = *(const v4f*)(B32 + off);
            v4f f1 = *(const v4f*)(B32 + off + 4);
            v8bf r;
            #pragma unroll
            for (int j = 0; j < 4; ++j) { r[j] = f2bf(f0[j]); r[4 + j] = f2bf(f1[j]); }
            return r;
        }
        return *(const v8bf*)(B16 + off);
    };
    auto stageB = [&](int buf, v8bf r) {
        if (LAYB == 0) {
            #pragma unroll
            for (int j = 0; j < 8; ++j) lsB[buf][bn_ + j][bk_] = r[j];
        } else {
            *(v8bf*)&lsB[buf][bn_][bk_] = r;
        }
    };

    // ---- wave tiling: wm in 0..3 (M), wn2 in 0..1 (N half) ----
    const int wave  = tid >> 5;
    const int lane  = tid & 31;
    const int wm    = wave >> 1;
    const int wn2   = wave & 1;
    const int l15   = lane & 15;
    const int mhalf = (lane >> 4) * 8;     // C/D rows 0..7 / 8..15 per lane half
    const int ka    = (lane >> 4) * 8;     // A frag k-halves {0..7,16..23}/{8..15,24..31}
    const int kb    = (lane >> 4) * 8;     // B frag k-halves {0..7,16..23}/{8..15,24..31}

    v8f acc0 = {}, acc1 = {};

    // ---- prologue: stage slab 0 ----
    if (LAYA == 0) {
        asyncStageA(0, 0);
    } else {
        v8bf ra = globA(0);
        stageA(0, ra);
    }
    {
        v8bf rb = globB(0);
        stageB(0, rb);
    }
    if (LAYA == 0) asm volatile("s_wait_asynccnt 0x0" ::: "memory");
    __syncthreads();

    const int nk = K / BK;
    for (int i = 0; i < nk; ++i) {
        const int cur = i & 1;

        v8bf ra, rb;
        if (i + 1 < nk) {
            if (LAYA == 0) asyncStageA(cur ^ 1, (i + 1) * BK);   // fire-and-forget
            else           ra = globA((i + 1) * BK);
            rb = globB((i + 1) * BK);
        }
        if (i + 2 < nk) {                       // uniform L2 prefetch, 2 steps ahead
            const int kp = (i + 2) * BK;
            const void* pa = (LAYA == 0) ? (const void*)(A + (long)(m0 + am) * K + (kp + ak))
                                         : (const void*)(A + (long)(kp + ak) * M + (m0 + am));
            __builtin_prefetch(pa, 0, 1);
            long off = (LAYB == 0) ? (long)(kp + bk_) * N + (n0 + bn_)
                                   : (long)(n0 + bn_) * K + (kp + bk_);
            __builtin_prefetch(BF32 ? (const void*)(B32 + off) : (const void*)(B16 + off), 0, 1);
        }

        union { v16bf v; v8bf h[2]; } fa, fb0, fb1;
        fa.h[0]  = *(const v8bf*)&lsA[cur][wm * 16 + l15][ka];
        fa.h[1]  = *(const v8bf*)&lsA[cur][wm * 16 + l15][ka + 16];
        fb0.h[0] = *(const v8bf*)&lsB[cur][wn2 * 32 + l15][kb];
        fb0.h[1] = *(const v8bf*)&lsB[cur][wn2 * 32 + l15][kb + 16];
        fb1.h[0] = *(const v8bf*)&lsB[cur][wn2 * 32 + 16 + l15][kb];
        fb1.h[1] = *(const v8bf*)&lsB[cur][wn2 * 32 + 16 + l15][kb + 16];

        acc0 = wmma_bf16(fa.v, fb0.v, acc0);
        acc1 = wmma_bf16(fa.v, fb1.v, acc1);

        if (i + 1 < nk) {
            if (LAYA == 1) stageA(cur ^ 1, ra);
            stageB(cur ^ 1, rb);
            if (LAYA == 0) asm volatile("s_wait_asynccnt 0x0" ::: "memory");
        }
        __syncthreads();
    }

    // ---- store wave tile: 16(M) x 32(N), 8 rows per lane half ----
    const int gmb = m0 + wm * 16 + mhalf;
    const int gn  = n0 + wn2 * 32 + l15;
    #pragma unroll
    for (int r = 0; r < 8; ++r) {
        long off = (long)bz * sC + (long)(gmb + r) * N + gn;
        float v0 = acc0[r] * scale;
        float v1 = acc1[r] * scale;
        if (OUTBF) { Cb[off] = f2bf(v0); Cb[off + 16] = f2bf(v1); }
        else       { Cf[off] = v0;       Cf[off + 16] = v1; }
    }
}

// ---------------------------------------------------------------------------
// f32 -> bf16 conversion (16B loads / 8B stores)
// ---------------------------------------------------------------------------
__global__ void cvt_f32_bf16(const float* __restrict__ x, __bf16* __restrict__ y, long n)
{
    long i = ((long)blockIdx.x * blockDim.x + threadIdx.x) * 4;
    long s = (long)gridDim.x * blockDim.x * 4;
    for (; i < n; i += s) {
        v4f f = *(const v4f*)(x + i);
        __bf16 o[4];
        #pragma unroll
        for (int j = 0; j < 4; ++j) o[j] = f2bf(f[j]);
        unsigned long long pk; __builtin_memcpy(&pk, o, 8);
        *(unsigned long long*)(y + i) = pk;
    }
}

// ---------------------------------------------------------------------------
// Row softmax, LDS-staged (one block per row of 2048 f32: 1 read + 1 write)
// ---------------------------------------------------------------------------
#define SMCOLS 2048

__global__ __launch_bounds__(256)
void softmax_rows(float* __restrict__ att)
{
    __shared__ float row[SMCOLS];
    __shared__ float red[256];
    float* p = att + (long)blockIdx.x * SMCOLS;
    const int tid = threadIdx.x;

    #pragma unroll
    for (int i = tid * 4; i < SMCOLS; i += 1024) *(v4f*)&row[i] = *(const v4f*)&p[i];
    __syncthreads();

    float mx = -3.0e38f;
    #pragma unroll
    for (int i = tid; i < SMCOLS; i += 256) mx = fmaxf(mx, row[i]);
    red[tid] = mx; __syncthreads();
    for (int s = 128; s > 0; s >>= 1) {
        if (tid < s) red[tid] = fmaxf(red[tid], red[tid + s]);
        __syncthreads();
    }
    mx = red[0]; __syncthreads();

    float sum = 0.0f;
    #pragma unroll
    for (int i = tid; i < SMCOLS; i += 256) {
        float e = __expf(row[i] - mx);
        row[i] = e;
        sum += e;
    }
    red[tid] = sum; __syncthreads();
    for (int s = 128; s > 0; s >>= 1) {
        if (tid < s) red[tid] += red[tid + s];
        __syncthreads();
    }
    const float inv = 1.0f / red[0];
    __syncthreads();

    #pragma unroll
    for (int i = tid * 4; i < SMCOLS; i += 1024) {
        v4f v = *(const v4f*)&row[i];
        v4f o = { v[0] * inv, v[1] * inv, v[2] * inv, v[3] * inv };
        *(v4f*)&p[i] = o;
    }
}

// ---------------------------------------------------------------------------
// BatchNorm stats + apply; out layout [B=16][C=512][N=2048]
// ---------------------------------------------------------------------------
#define NB   16
#define NC   512
#define NPOS 2048

__global__ __launch_bounds__(256)
void bn_stats(const float* __restrict__ out, float* __restrict__ meanv,
              float* __restrict__ istdv)
{
    const int c = blockIdx.x, tid = threadIdx.x;
    __shared__ float rs[256], rs2[256];
    float s = 0.0f, s2 = 0.0f;
    for (int b = 0; b < NB; ++b) {
        const float* p = out + ((long)b * NC + c) * NPOS;
        for (int i = tid * 4; i < NPOS; i += 1024) {
            v4f v = *(const v4f*)&p[i];
            #pragma unroll
            for (int j = 0; j < 4; ++j) { s += v[j]; s2 += v[j] * v[j]; }
        }
    }
    rs[tid] = s; rs2[tid] = s2; __syncthreads();
    for (int t = 128; t > 0; t >>= 1) {
        if (tid < t) { rs[tid] += rs[tid + t]; rs2[tid] += rs2[tid + t]; }
        __syncthreads();
    }
    if (tid == 0) {
        float inv_n = 1.0f / (float)(NB * NPOS);
        float mean = rs[0] * inv_n;
        float var  = rs2[0] * inv_n - mean * mean;
        meanv[c] = mean;
        istdv[c] = rsqrtf(var + 1e-5f);
    }
}

__global__ void bn_apply(float* __restrict__ out, const float* __restrict__ meanv,
                         const float* __restrict__ istdv, const float* __restrict__ gamma,
                         const float* __restrict__ beta, long n)
{
    long i = ((long)blockIdx.x * blockDim.x + threadIdx.x) * 4;
    long s = (long)gridDim.x * blockDim.x * 4;
    for (; i < n; i += s) {
        int c = (int)((i >> 11) & (NC - 1));          // [B][C][2048], 4-chunks stay in-channel
        float sc = istdv[c] * gamma[c];
        float sh = beta[c] - meanv[c] * sc;
        v4f v = *(const v4f*)(out + i);
        v4f o = { v[0] * sc + sh, v[1] * sc + sh, v[2] * sc + sh, v[3] * sc + sh };
        *(v4f*)(out + i) = o;
    }
}

// ---------------------------------------------------------------------------
// Launch
// ---------------------------------------------------------------------------
extern "C" void kernel_launch(void* const* d_in, const int* in_sizes, int n_in,
                              void* d_out, int out_size, void* d_ws, size_t ws_size,
                              hipStream_t stream)
{
    (void)in_sizes; (void)n_in; (void)out_size; (void)ws_size;

    const int B = 16, C = 512, E = 256, An = 2048, Bn = 2048;

    const float* inA   = (const float*)d_in[0];
    const float* inB   = (const float*)d_in[1];
    const float* wA    = (const float*)d_in[2];
    const float* wB    = (const float*)d_in[3];
    const float* wBh   = (const float*)d_in[4];
    const float* wOut  = (const float*)d_in[5];
    const float* gamma = (const float*)d_in[6];
    const float* beta  = (const float*)d_in[7];

    float* out = (float*)d_out;                      // [16,512,2048]
    float* att = out + (long)B * C * An;             // [16,2048,2048]

    // ---- workspace carve-out ----
    char* wp = (char*)d_ws;
    auto carve = [&](size_t bytes) -> char* {
        char* r = wp; wp += (bytes + 255) & ~(size_t)255; return r;
    };
    __bf16* iA16  = (__bf16*)carve((size_t)B * C * An * 2);
    __bf16* iB16  = (__bf16*)carve((size_t)B * C * Bn * 2);
    __bf16* wA16  = (__bf16*)carve((size_t)E * C * 2);
    __bf16* wB16  = (__bf16*)carve((size_t)E * C * 2);
    __bf16* wBh16 = (__bf16*)carve((size_t)E * C * 2);
    __bf16* wO16  = (__bf16*)carve((size_t)C * E * 2);
    __bf16* pA    = (__bf16*)carve((size_t)B * E * An * 2);
    __bf16* pB    = (__bf16*)carve((size_t)B * E * Bn * 2);
    __bf16* pBh   = (__bf16*)carve((size_t)B * E * Bn * 2);
    __bf16* mid   = (__bf16*)carve((size_t)B * E * An * 2);
    float*  meanv = (float*)carve((size_t)C * 4);
    float*  istdv = (float*)carve((size_t)C * 4);

    // ---- 1) cast inputs/weights to bf16 ----
    cvt_f32_bf16<<<8192, 256, 0, stream>>>(inA, iA16, (long)B * C * An);
    cvt_f32_bf16<<<8192, 256, 0, stream>>>(inB, iB16, (long)B * C * Bn);
    cvt_f32_bf16<<<128,  256, 0, stream>>>(wA,  wA16,  (long)E * C);
    cvt_f32_bf16<<<128,  256, 0, stream>>>(wB,  wB16,  (long)E * C);
    cvt_f32_bf16<<<128,  256, 0, stream>>>(wBh, wBh16, (long)E * C);
    cvt_f32_bf16<<<128,  256, 0, stream>>>(wOut, wO16, (long)C * E);

    dim3 blk(TBLK);

    // ---- 2) projections: [E,C] x [C,N] -> bf16 [E,N] per batch ----
    {
        dim3 g(An / BN, E / BM, B);
        gemm_wmma_bf16<0, 0, false, true><<<g, blk, 0, stream>>>(
            wA16, iA16, nullptr, pA, E, An, C, 0, (long)C * An, (long)E * An, 1.0f);
        gemm_wmma_bf16<0, 0, false, true><<<g, blk, 0, stream>>>(
            wB16, iB16, nullptr, pB, E, Bn, C, 0, (long)C * Bn, (long)E * Bn, 1.0f);
        gemm_wmma_bf16<0, 0, false, true><<<g, blk, 0, stream>>>(
            wBh16, iB16, nullptr, pBh, E, Bn, C, 0, (long)C * Bn, (long)E * Bn, 1.0f);
    }

    // ---- 3) h = (A^T x Bp) / 16 : M=An, N=Bn, K=E ; A is [E][An] => KM ----
    {
        dim3 g(Bn / BN, An / BM, B);
        gemm_wmma_bf16<1, 0, false, false><<<g, blk, 0, stream>>>(
            pA, pB, att, nullptr, An, Bn, E,
            (long)E * An, (long)E * Bn, (long)An * Bn, 1.0f / 16.0f);
    }

    // ---- 4) softmax over Bn, in place in d_out att region ----
    softmax_rows<<<B * An, 256, 0, stream>>>(att);

    // ---- 5) out_mid = Bh x att^T : M=E, N=An, K=Bn ; att [An][Bn] => NK (f32) ----
    {
        dim3 g(An / BN, E / BM, B);
        gemm_wmma_bf16<0, 1, true, true><<<g, blk, 0, stream>>>(
            pBh, att, nullptr, mid, E, An, Bn,
            (long)E * Bn, (long)An * Bn, (long)E * An, 1.0f);
    }

    // ---- 6) out_pre = W_out x out_mid : M=C, N=An, K=E -> f32 into d_out ----
    {
        dim3 g(An / BN, C / BM, B);
        gemm_wmma_bf16<0, 0, false, false><<<g, blk, 0, stream>>>(
            wO16, mid, out, nullptr, C, An, E, 0, (long)E * An, (long)C * An, 1.0f);
    }

    // ---- 7) BatchNorm (training mode) in place on d_out out-region ----
    bn_stats<<<C, 256, 0, stream>>>(out, meanv, istdv);
    bn_apply<<<8192, 256, 0, stream>>>(out, meanv, istdv, gamma, beta, (long)B * C * An);
}
